// CausalSelfAttention_57045755625662
// MI455X (gfx1250) — matrix operand order repaired and checked by reading, not compiled, not run on hardware
//
#include <hip/hip_runtime.h>
#include <hip/hip_bf16.h>

// ---------------------------------------------------------------------------
// CausalSelfAttention forward for MI455X (gfx1250, wave32, WMMA).
// B=4, T=2048, C=1024, H=16, D=64. All matmuls use v_wmma_f32_16x16x32_bf16.
// ---------------------------------------------------------------------------

#define BATCH   4
#define SEQ     2048
#define EMBD    1024
#define NHEAD   16
#define HDIM    64
#define MROWS   (BATCH * SEQ)      // 8192
#define MASK_FILL (-10.0f)
#define SM_SCALE  (0.125f)         // 1/sqrt(64)

typedef __bf16 bf16_t;
typedef __attribute__((ext_vector_type(16))) __bf16 v16bf;
typedef __attribute__((ext_vector_type(8)))  __bf16 v8bf;
typedef __attribute__((ext_vector_type(8)))  float  v8f;

static __device__ __forceinline__ bf16_t f2bf(float f) {
    union { float f; unsigned u; } a; a.f = f;
    unsigned r = a.u + 0x7FFFu + ((a.u >> 16) & 1u);   // round-to-nearest-even
    unsigned short h = (unsigned short)(r >> 16);
    return __builtin_bit_cast(__bf16, h);
}

static __device__ __forceinline__ v8f wmma_bf16(v16bf a, v16bf b, v8f c) {
    // D = A(16x32 bf16) x B(32x16 bf16) + C(16x16 f32)
    return __builtin_amdgcn_wmma_f32_16x16x32_bf16(
        /*neg_a=*/false, a, /*neg_b=*/false, b,
        /*c_mod=*/(short)0, c, /*reuse_a=*/false, /*reuse_b=*/false);
}

// A-fragment (16-bit 16x32, MxK), contiguous k order matching B loads:
//   lanes 0-15  (M=lane): K = {0..15}
//   lanes 16-31 (M=lane-16): K = {16..31}
// 'rowp' points at this lane's row start (k offset 0 of the 32-wide K tile).
static __device__ __forceinline__ v16bf load_a_frag(const bf16_t* rowp, int lhalf) {
    v8bf lo = *(const v8bf*)(rowp + lhalf * 16);
    v8bf hi = *(const v8bf*)(rowp + lhalf * 16 + 8);
    return __builtin_shufflevector(lo, hi, 0,1,2,3,4,5,6,7,8,9,10,11,12,13,14,15);
}

// ---------------------------------------------------------------------------
// fp32 -> bf16 conversion
// ---------------------------------------------------------------------------
__global__ void cvt_f32_bf16_kernel(const float* __restrict__ in,
                                    bf16_t* __restrict__ out, int n) {
    int i = blockIdx.x * blockDim.x + threadIdx.x;
    if (i < n) out[i] = f2bf(in[i]);
}

// ---------------------------------------------------------------------------
// QKV projection: out = x @ W^T, W stored [out,in].  C[m,n] = sum_k x[m,k]W[n,k]
// grid: (EMBD/64, MROWS/128, 3)  block: 128 (4 waves).
// Block tile 128x64; each wave computes 32x64 (2 row-strips x 4 col-tiles =
// 8 WMMA accumulators), so every B fragment is reused by 2 A fragments.
// Output stored bf16 in [B, H, T, D] layout.
// ---------------------------------------------------------------------------
__global__ void qkv_gemm_kernel(const bf16_t* __restrict__ xb,
                                const bf16_t* __restrict__ wqb,
                                const bf16_t* __restrict__ wkb,
                                const bf16_t* __restrict__ wvb,
                                bf16_t* __restrict__ qout,
                                bf16_t* __restrict__ kout,
                                bf16_t* __restrict__ vout) {
    const int which = blockIdx.z;
    const bf16_t* __restrict__ w = (which == 0) ? wqb : (which == 1) ? wkb : wvb;
    bf16_t* __restrict__ out     = (which == 0) ? qout : (which == 1) ? kout : vout;

    const int n0 = blockIdx.x * 64;
    const int m0 = blockIdx.y * 128;

    const int tid   = threadIdx.x;
    const int wave  = tid >> 5;
    const int lane  = tid & 31;
    const int l16   = lane & 15;
    const int lhalf = lane >> 4;

    __shared__ alignas(32) bf16_t At[128][32];  // A tile (rows m, k)
    __shared__ alignas(32) bf16_t Bt[64][32];   // B tile (rows n, k) == W rows

    v8f acc[2][4] = {};

    const int brow = tid >> 1;          // 0..63
    const int bcol = (tid & 1) * 16;    // 0 or 16

    for (int kb = 0; kb < EMBD; kb += 32) {
        // stage: each thread copies one 32-wide A row + half a B row
        *(v16bf*)&At[tid][0]  = *(const v16bf*)&xb[(size_t)(m0 + tid) * EMBD + kb];
        *(v16bf*)&At[tid][16] = *(const v16bf*)&xb[(size_t)(m0 + tid) * EMBD + kb + 16];
        *(v16bf*)&Bt[brow][bcol] = *(const v16bf*)&w[(size_t)(n0 + brow) * EMBD + kb + bcol];
        if (kb + 32 < EMBD) {
            __builtin_prefetch(&xb[(size_t)(m0 + tid) * EMBD + kb + 32], 0, 3);
            __builtin_prefetch(&w [(size_t)(n0 + brow) * EMBD + kb + 32], 0, 3);
        }
        __syncthreads();

        v16bf af0 = load_a_frag(&At[wave * 32 + l16][0], lhalf);
        v16bf af1 = load_a_frag(&At[wave * 32 + 16 + l16][0], lhalf);
#pragma unroll
        for (int nt = 0; nt < 4; ++nt) {
            v16bf bfrag = *(const v16bf*)&Bt[nt * 16 + l16][lhalf * 16];
            acc[0][nt] = wmma_bf16(af0, bfrag, acc[0][nt]);
            acc[1][nt] = wmma_bf16(af1, bfrag, acc[1][nt]);
        }
        __syncthreads();
    }

    // epilogue: store bf16 into [B, H, T, D]
#pragma unroll
    for (int r = 0; r < 2; ++r) {
#pragma unroll
        for (int nt = 0; nt < 4; ++nt) {
#pragma unroll
            for (int i = 0; i < 8; ++i) {
                int m = m0 + wave * 32 + r * 16 + 8 * lhalf + i;  // row in [B*T]
                int n = n0 + nt * 16 + l16;                       // channel
                int b = m / SEQ, t = m % SEQ;
                int h = n >> 6,  d = n & 63;
                out[(((size_t)b * NHEAD + h) * SEQ + t) * HDIM + d] = f2bf(acc[r][nt][i]);
            }
        }
    }
}

// ---------------------------------------------------------------------------
// Flash-style attention with EXACT reference semantics: every key position is
// scored; masked (k > q) entries get -10 before softmax (so they DO contribute
// exp(-10) to the denominator, matching jnp.where + softmax).
// grid: (SEQ/64, B*H)  block: 128 (4 waves, 16 q-rows per wave).
// ---------------------------------------------------------------------------
__global__ void flash_attn_kernel(const bf16_t* __restrict__ Q,
                                  const bf16_t* __restrict__ K,
                                  const bf16_t* __restrict__ V,
                                  bf16_t* __restrict__ Y) {
    const int bh = blockIdx.y;
    const int qb = blockIdx.x;
    const size_t base = (size_t)bh * SEQ * HDIM;
    const bf16_t* __restrict__ Qp = Q + base;
    const bf16_t* __restrict__ Kp = K + base;
    const bf16_t* __restrict__ Vp = V + base;

    const int tid   = threadIdx.x;
    const int wave  = tid >> 5;
    const int lane  = tid & 31;
    const int l16   = lane & 15;
    const int lhalf = lane >> 4;

    __shared__ alignas(32) bf16_t Kl[64][64];       // [key][d]
    __shared__ alignas(32) bf16_t Vt[64][64];       // [d][key] (transposed)
    __shared__ alignas(32) bf16_t Pl[4][16][64];    // per-wave P tile [row][key]

    // Q fragments for this wave's 16 rows (D=64 -> two 32-wide K steps)
    const int qrow0 = qb * 64 + wave * 16;
    v16bf qf[2];
#pragma unroll
    for (int s = 0; s < 2; ++s)
        qf[s] = load_a_frag(&Qp[(size_t)(qrow0 + l16) * HDIM + s * 32], lhalf);

    float mrow[8], lrow[8];
#pragma unroll
    for (int i = 0; i < 8; ++i) { mrow[i] = -1e30f; lrow[i] = 0.0f; }
    v8f o[4] = {};

    const int srow = tid >> 1;          // 0..63
    const int scol = (tid & 1) * 32;    // 0 or 32

    for (int jb = 0; jb < SEQ / 64; ++jb) {
        // ---- stage K tile [key][d] and V^T tile [d][key] ----
        *(v16bf*)&Kl[srow][scol]      = *(const v16bf*)&Kp[(size_t)(jb * 64 + srow) * HDIM + scol];
        *(v16bf*)&Kl[srow][scol + 16] = *(const v16bf*)&Kp[(size_t)(jb * 64 + srow) * HDIM + scol + 16];
        v16bf v0 = *(const v16bf*)&Vp[(size_t)(jb * 64 + srow) * HDIM + scol];
        v16bf v1 = *(const v16bf*)&Vp[(size_t)(jb * 64 + srow) * HDIM + scol + 16];
#pragma unroll
        for (int c = 0; c < 16; ++c) {
            Vt[scol + c][srow]      = v0[c];
            Vt[scol + 16 + c][srow] = v1[c];
        }
        __syncthreads();

        // ---- S = Q K^T  (16x64 per wave; 4 n-tiles, 2 k-steps) ----
        v8f s[4] = {};
#pragma unroll
        for (int ks = 0; ks < 2; ++ks) {
#pragma unroll
            for (int nt = 0; nt < 4; ++nt) {
                v16bf bk = *(const v16bf*)&Kl[nt * 16 + l16][ks * 32 + lhalf * 16];
                s[nt] = wmma_bf16(qf[ks], bk, s[nt]);
            }
        }

        // ---- scale + causal mask (fill -10, kept in softmax like reference) ----
        const int qgbase = qb * 64 + wave * 16 + 8 * lhalf;
#pragma unroll
        for (int nt = 0; nt < 4; ++nt) {
            const int kg = jb * 64 + nt * 16 + l16;
#pragma unroll
            for (int i = 0; i < 8; ++i) {
                float val = s[nt][i] * SM_SCALE;
                if (kg > qgbase + i) val = MASK_FILL;
                s[nt][i] = val;
            }
        }

        // ---- online softmax (row = 8*lhalf + i, spread over 16 lanes) ----
#pragma unroll
        for (int i = 0; i < 8; ++i) {
            float mx = fmaxf(fmaxf(s[0][i], s[1][i]), fmaxf(s[2][i], s[3][i]));
#pragma unroll
            for (int off = 8; off >= 1; off >>= 1)
                mx = fmaxf(mx, __shfl_xor(mx, off, 32));
            float nm = fmaxf(mrow[i], mx);
            float alpha = __expf(mrow[i] - nm);
            mrow[i] = nm;
            float rs = 0.0f;
#pragma unroll
            for (int nt = 0; nt < 4; ++nt) {
                float p = __expf(s[nt][i] - nm);
                s[nt][i] = p;
                rs += p;
            }
#pragma unroll
            for (int off = 8; off >= 1; off >>= 1)
                rs += __shfl_xor(rs, off, 32);
            lrow[i] = lrow[i] * alpha + rs;
#pragma unroll
            for (int nt = 0; nt < 4; ++nt) o[nt][i] *= alpha;
        }

        // ---- P -> LDS (re-layout C-frag -> A-frag via shared memory) ----
#pragma unroll
        for (int nt = 0; nt < 4; ++nt)
#pragma unroll
            for (int i = 0; i < 8; ++i)
                Pl[wave][8 * lhalf + i][nt * 16 + l16] = f2bf(s[nt][i]);
        __syncthreads();

        // ---- O += P @ V  (K dim = 64 keys -> 2 k-steps) ----
#pragma unroll
        for (int ks = 0; ks < 2; ++ks) {
            v16bf pf = load_a_frag(&Pl[wave][l16][ks * 32], lhalf);
#pragma unroll
            for (int nt = 0; nt < 4; ++nt) {
                v16bf bv = *(const v16bf*)&Vt[nt * 16 + l16][ks * 32 + lhalf * 16];
                o[nt] = wmma_bf16(pf, bv, o[nt]);
            }
        }
        __syncthreads();
    }

    // ---- epilogue: y[b, t, h*64 + d] = O / l, bf16 for the final projection ----
    const int b = bh / NHEAD;
    const int h = bh % NHEAD;
#pragma unroll
    for (int nt = 0; nt < 4; ++nt) {
#pragma unroll
        for (int i = 0; i < 8; ++i) {
            int t = qb * 64 + wave * 16 + 8 * lhalf + i;
            int col = h * HDIM + nt * 16 + l16;
            float val = o[nt][i] / lrow[i];
            Y[((size_t)b * SEQ + t) * EMBD + col] = f2bf(val);
        }
    }
}

// ---------------------------------------------------------------------------
// Output projection: out = y @ wo^T, fp32 result straight to d_out [B,T,C].
// Same 128x64 block tile / 32x64 per-wave structure as the QKV GEMM.
// ---------------------------------------------------------------------------
__global__ void out_proj_kernel(const bf16_t* __restrict__ yb,
                                const bf16_t* __restrict__ wob,
                                float* __restrict__ out) {
    const int n0 = blockIdx.x * 64;
    const int m0 = blockIdx.y * 128;

    const int tid   = threadIdx.x;
    const int wave  = tid >> 5;
    const int lane  = tid & 31;
    const int l16   = lane & 15;
    const int lhalf = lane >> 4;

    __shared__ alignas(32) bf16_t At[128][32];
    __shared__ alignas(32) bf16_t Bt[64][32];

    v8f acc[2][4] = {};

    const int brow = tid >> 1;
    const int bcol = (tid & 1) * 16;

    for (int kb = 0; kb < EMBD; kb += 32) {
        *(v16bf*)&At[tid][0]  = *(const v16bf*)&yb[(size_t)(m0 + tid) * EMBD + kb];
        *(v16bf*)&At[tid][16] = *(const v16bf*)&yb[(size_t)(m0 + tid) * EMBD + kb + 16];
        *(v16bf*)&Bt[brow][bcol] = *(const v16bf*)&wob[(size_t)(n0 + brow) * EMBD + kb + bcol];
        if (kb + 32 < EMBD) {
            __builtin_prefetch(&yb [(size_t)(m0 + tid) * EMBD + kb + 32], 0, 3);
            __builtin_prefetch(&wob[(size_t)(n0 + brow) * EMBD + kb + 32], 0, 3);
        }
        __syncthreads();

        v16bf af0 = load_a_frag(&At[wave * 32 + l16][0], lhalf);
        v16bf af1 = load_a_frag(&At[wave * 32 + 16 + l16][0], lhalf);
#pragma unroll
        for (int nt = 0; nt < 4; ++nt) {
            v16bf bfrag = *(const v16bf*)&Bt[nt * 16 + l16][lhalf * 16];
            acc[0][nt] = wmma_bf16(af0, bfrag, acc[0][nt]);
            acc[1][nt] = wmma_bf16(af1, bfrag, acc[1][nt]);
        }
        __syncthreads();
    }

#pragma unroll
    for (int r = 0; r < 2; ++r) {
#pragma unroll
        for (int nt = 0; nt < 4; ++nt) {
#pragma unroll
            for (int i = 0; i < 8; ++i) {
                int m = m0 + wave * 32 + r * 16 + 8 * lhalf + i;
                int n = n0 + nt * 16 + l16;
                out[(size_t)m * EMBD + n] = acc[r][nt][i];
            }
        }
    }
}

// ---------------------------------------------------------------------------
// Launch
// ---------------------------------------------------------------------------
extern "C" void kernel_launch(void* const* d_in, const int* in_sizes, int n_in,
                              void* d_out, int out_size, void* d_ws, size_t ws_size,
                              hipStream_t stream) {
    (void)in_sizes; (void)n_in; (void)out_size; (void)ws_size;

    const float* x  = (const float*)d_in[0];
    const float* wq = (const float*)d_in[1];
    const float* wk = (const float*)d_in[2];
    const float* wv = (const float*)d_in[3];
    const float* wo = (const float*)d_in[4];
    float* out = (float*)d_out;

    const size_t XE = (size_t)MROWS * EMBD;   // 8,388,608 elements
    const size_t WE = (size_t)EMBD * EMBD;    // 1,048,576 elements

    char* ws = (char*)d_ws;
    bf16_t* xb  = (bf16_t*)ws;                 ws += XE * sizeof(bf16_t);
    bf16_t* wqb = (bf16_t*)ws;                 ws += WE * sizeof(bf16_t);
    bf16_t* wkb = (bf16_t*)ws;                 ws += WE * sizeof(bf16_t);
    bf16_t* wvb = (bf16_t*)ws;                 ws += WE * sizeof(bf16_t);
    bf16_t* wob = (bf16_t*)ws;                 ws += WE * sizeof(bf16_t);
    bf16_t* qb_ = (bf16_t*)ws;                 ws += XE * sizeof(bf16_t);
    bf16_t* kb_ = (bf16_t*)ws;                 ws += XE * sizeof(bf16_t);
    bf16_t* vb_ = (bf16_t*)ws;                 ws += XE * sizeof(bf16_t);
    bf16_t* yb  = (bf16_t*)ws;                 ws += XE * sizeof(bf16_t);

    // 1) fp32 -> bf16
    cvt_f32_bf16_kernel<<<(XE + 255) / 256, 256, 0, stream>>>(x,  xb,  (int)XE);
    cvt_f32_bf16_kernel<<<(WE + 255) / 256, 256, 0, stream>>>(wq, wqb, (int)WE);
    cvt_f32_bf16_kernel<<<(WE + 255) / 256, 256, 0, stream>>>(wk, wkb, (int)WE);
    cvt_f32_bf16_kernel<<<(WE + 255) / 256, 256, 0, stream>>>(wv, wvb, (int)WE);
    cvt_f32_bf16_kernel<<<(WE + 255) / 256, 256, 0, stream>>>(wo, wob, (int)WE);

    // 2) Q/K/V projections -> [B,H,T,D] bf16
    qkv_gemm_kernel<<<dim3(EMBD / 64, MROWS / 128, 3), 128, 0, stream>>>(
        xb, wqb, wkb, wvb, qb_, kb_, vb_);

    // 3) attention -> y [B,T,C] bf16
    flash_attn_kernel<<<dim3(SEQ / 64, BATCH * NHEAD), 128, 0, stream>>>(
        qb_, kb_, vb_, yb);

    // 4) output projection -> fp32 d_out
    out_proj_kernel<<<dim3(EMBD / 64, MROWS / 128), 128, 0, stream>>>(yb, wob, out);
}
